// EncoderLayer_hybrid2_33698313404912
// MI455X (gfx1250) — matrix operand order repaired and checked by reading, compiled, not run on hardware
//
#include <hip/hip_runtime.h>
#include <hip/hip_bf16.h>
#include <math.h>

typedef _Float16 v16h __attribute__((ext_vector_type(16)));
typedef _Float16 v8h  __attribute__((ext_vector_type(8)));
typedef float    v8f  __attribute__((ext_vector_type(8)));

#define BB   4
#define SS   2048
#define EE   512
#define HH   8
#define DD   64
#define FFF  2048
#define NTOK (BB*SS)

// LDS tile: 64 columns x 32 K-halves, padded row stride of 40 halves (80B).
// Fragment reads stride 80B = 20 dwords; gcd(20,64)=4 -> 16 lanes hit 16
// distinct banks (conflict-free).
#define LPAD 40

// ---------------------------------------------------------------------------
// A-fragment loader for V_WMMA_F32_16X16X32_F16 (ISA 7.12.2):
// lanes 0-15 row M=lane, K={0..7,16..23}; lanes 16-31 row M=lane-16,
// K={8..15,24..31}.  g = lane>>4.  Two contiguous 16B runs per lane.
static __device__ __forceinline__ v16h load_a_frag(const _Float16* __restrict__ row,
                                                   int kc, int g) {
  v8h lo = *(const v8h*)(row + kc + 8 * g);
  v8h hi = *(const v8h*)(row + kc + 16 + 8 * g);
  v16h r;
#pragma unroll
  for (int i = 0; i < 8; ++i) { r[i] = lo[i]; r[8 + i] = hi[i]; }
  return r;
}

// B fragment from an LDS-staged K-major column (32 K-halves, padded stride):
// lane l holds col N=l%16, elements e=0..7 are K = 8*g + e, e=8..15 are K = 16 + 8*g + (e-8).
static __device__ __forceinline__ v16h load_b_frag_lds(const _Float16* colbase, int g) {
  v8h lo = *(const v8h*)(colbase + 8 * g);
  v8h hi = *(const v8h*)(colbase + 16 + 8 * g);
  v16h r;
#pragma unroll
  for (int i = 0; i < 8; ++i) { r[i] = lo[i]; r[8 + i] = hi[i]; }
  return r;
}

static __device__ __forceinline__ v8f wmma_f16(v16h a, v16h b, v8f c) {
  return __builtin_amdgcn_wmma_f32_16x16x32_f16(false, a, false, b, (short)0, c,
                                                false, false);
}

// Cooperative stage of a 64-col x 32-K f16 chunk into padded LDS.
// 128 threads: thread t stages col n=t>>1, half hh=t&1 (16 halves = 32B).
static __device__ __forceinline__ void stage_chunk(_Float16* sbuf,
                                                   const _Float16* __restrict__ gsrc,
                                                   size_t col_stride, int kc) {
  const int n = threadIdx.x >> 1, hh = threadIdx.x & 1;
  v16h v = *(const v16h*)(gsrc + (size_t)n * col_stride + kc + 16 * hh);
  _Float16* dst = sbuf + n * LPAD + 16 * hh;
  v8h lo, hi;
#pragma unroll
  for (int i = 0; i < 8; ++i) { lo[i] = v[i]; hi[i] = v[8 + i]; }
  *(v8h*)(dst) = lo;
  *(v8h*)(dst + 8) = hi;
}

// ---------------------------------------------------------------------------
// Kernel 1: transpose+convert MLP weights to f16.
__global__ void convert_weights(const float* __restrict__ W1, const float* __restrict__ W2,
                                _Float16* __restrict__ W1t, _Float16* __restrict__ W2t) {
  int idx = blockIdx.x * 256 + threadIdx.x;            // 0 .. EE*FFF-1
  if (idx < EE * FFF) {
    int n1 = idx / EE, k1 = idx % EE;                  // W1t[n,k] = W1[k,n]
    W1t[idx] = (_Float16)W1[(size_t)k1 * FFF + n1];
    int n2 = idx / FFF, k2 = idx % FFF;                // W2t[n,k] = W2[k,n]
    W2t[idx] = (_Float16)W2[(size_t)k2 * EE + n2];
  }
}

// ---------------------------------------------------------------------------
// Kernel 2: LayerNorm + per-head V projection (stored transposed, f16) and
// scalar Q/K projections.  One block (256 threads) per token.
__global__ void ln_qkv_kernel(const float* __restrict__ x,
                              const float* __restrict__ Wv, const float* __restrict__ bv,
                              const float* __restrict__ Wq, const float* __restrict__ bq,
                              const float* __restrict__ Wk, const float* __restrict__ bk,
                              float* __restrict__ Qout, float* __restrict__ Kout,
                              _Float16* __restrict__ Vt) {
  __shared__ float sx[EE];
  __shared__ float red[256];
  const int tok = blockIdx.x;
  const int tid = threadIdx.x;
  const float x0 = x[(size_t)tok * EE + tid];
  const float x1 = x[(size_t)tok * EE + tid + 256];

  red[tid] = x0 + x1;
  __syncthreads();
#pragma unroll
  for (int off = 128; off > 0; off >>= 1) {
    if (tid < off) red[tid] += red[tid + off];
    __syncthreads();
  }
  const float mean = red[0] * (1.0f / EE);
  __syncthreads();
  const float d0 = x0 - mean, d1 = x1 - mean;
  red[tid] = d0 * d0 + d1 * d1;
  __syncthreads();
#pragma unroll
  for (int off = 128; off > 0; off >>= 1) {
    if (tid < off) red[tid] += red[tid + off];
    __syncthreads();
  }
  const float rstd = rsqrtf(red[0] * (1.0f / EE) + 1e-5f);
  sx[tid] = d0 * rstd;
  sx[tid + 256] = d1 * rstd;
  __syncthreads();

  const int h = tid >> 5, lane = tid & 31;
  const float* xh = sx + h * DD;
  const int b = tok / SS, s = tok % SS;
  const int bh = b * HH + h;

  // V[e] = sum_d xh[d] * Wv[h,d,e] + bv[h,e]; each lane owns e=lane, lane+32.
  float a0 = 0.f, a1 = 0.f;
#pragma unroll 8
  for (int d = 0; d < DD; ++d) {
    const float xv = xh[d];
    const float* wr = Wv + ((size_t)(h * DD + d)) * DD;
    a0 = fmaf(xv, wr[lane], a0);
    a1 = fmaf(xv, wr[lane + 32], a1);
  }
  a0 += bv[h * DD + lane];
  a1 += bv[h * DD + lane + 32];
  Vt[((size_t)bh * DD + lane) * SS + s]      = (_Float16)a0;
  Vt[((size_t)bh * DD + lane + 32) * SS + s] = (_Float16)a1;

  // Scalar Q / K projections: wave-level dot product.
  float qp = xh[lane] * Wq[h * DD + lane] + xh[lane + 32] * Wq[h * DD + lane + 32];
  float kp = xh[lane] * Wk[h * DD + lane] + xh[lane + 32] * Wk[h * DD + lane + 32];
#pragma unroll
  for (int m = 16; m > 0; m >>= 1) {
    qp += __shfl_xor(qp, m, 32);
    kp += __shfl_xor(kp, m, 32);
  }
  if (lane == 0) {
    Qout[(size_t)bh * SS + s] = qp + bq[h];
    Kout[(size_t)bh * SS + s] = kp + bk[h];
  }
}

// ---------------------------------------------------------------------------
// Kernel 3: flash attention with scalar quadratic-distance scores.
// Grid: (S/64 query tiles, B*H).  Block: 128 threads = 4 waves; each wave owns
// a 16-query tile.  The 32x64 f16 V chunk is staged once per block in LDS
// (4x less L2 traffic than per-wave loads); online softmax in base-2.
__global__ void attn_kernel(const float* __restrict__ Qg, const float* __restrict__ Kg,
                            const _Float16* __restrict__ Vt,
                            _Float16* __restrict__ head_out) {
  __shared__ __align__(32) _Float16 vbuf[64 * LPAD];
  const int bh = blockIdx.y;
  const int wave = threadIdx.x >> 5, lane = threadIdx.x & 31;
  const int q0 = blockIdx.x * 64 + wave * 16;
  const int m = lane & 15, g = lane >> 4;

  const float* Kp = Kg + (size_t)bh * SS;
  const _Float16* Vb = Vt + (size_t)bh * DD * SS;
  const float qv = Qg[(size_t)bh * SS + q0 + m];
  const float c2 = 0.125f * 1.44269504088896340736f;   // (1/sqrt(D)) * log2(e)

  float mrow = -INFINITY, lrow = 0.0f;
  v8f acc[4];
#pragma unroll
  for (int t = 0; t < 4; ++t)
#pragma unroll
    for (int r = 0; r < 8; ++r) acc[t][r] = 0.0f;

  for (int kc = 0; kc < SS; kc += 32) {
    stage_chunk(vbuf, Vb, SS, kc);          // V chunk [64 cols][32 K] -> LDS
    __syncthreads();

    // K scalars in exactly the A-fragment element order for this lane.
    v8f k0 = *(const v8f*)(Kp + kc + 8 * g);
    v8f k1 = *(const v8f*)(Kp + kc + 16 + 8 * g);
    float sc[16];
#pragma unroll
    for (int i = 0; i < 8; ++i) { float d = qv - k0[i]; sc[i] = -c2 * d * d; }
#pragma unroll
    for (int i = 0; i < 8; ++i) { float d = qv - k1[i]; sc[8 + i] = -c2 * d * d; }

    float cmax = sc[0];
#pragma unroll
    for (int i = 1; i < 16; ++i) cmax = fmaxf(cmax, sc[i]);
    cmax = fmaxf(cmax, __shfl_xor(cmax, 16, 32));      // combine row halves
    const float mnew = fmaxf(mrow, cmax);
    const float alpha = exp2f(mrow - mnew);

    float p[16], rs = 0.0f;
#pragma unroll
    for (int i = 0; i < 16; ++i) { p[i] = exp2f(sc[i] - mnew); rs += p[i]; }
    rs += __shfl_xor(rs, 16, 32);
    lrow = lrow * alpha + rs;
    mrow = mnew;

    // Broadcast per-row alpha to the C-layout (lane needs rows r+8g).
    float af[8];
#pragma unroll
    for (int r = 0; r < 8; ++r) af[r] = __shfl(alpha, r + 8 * g, 32);
#pragma unroll
    for (int t = 0; t < 4; ++t)
#pragma unroll
      for (int r = 0; r < 8; ++r) acc[t][r] *= af[r];

    v16h a;
#pragma unroll
    for (int i = 0; i < 16; ++i) a[i] = (_Float16)p[i];

#pragma unroll
    for (int t = 0; t < 4; ++t) {
      v16h bf = load_b_frag_lds(vbuf + (16 * t + m) * LPAD, g);
      acc[t] = wmma_f16(a, bf, acc[t]);
    }
    __syncthreads();                        // protect LDS before next stage
  }

  const float linv = 1.0f / lrow;
  float lf[8];
#pragma unroll
  for (int r = 0; r < 8; ++r) lf[r] = __shfl(linv, r + 8 * g, 32);

  const int b = bh >> 3, h = bh & 7;
#pragma unroll
  for (int t = 0; t < 4; ++t)
#pragma unroll
    for (int r = 0; r < 8; ++r) {
      const int row = q0 + r + 8 * g;
      head_out[((size_t)(b * SS + row)) * EE + h * DD + 16 * t + m] =
          (_Float16)(acc[t][r] * lf[r]);
    }
}

// ---------------------------------------------------------------------------
// Generic WMMA GEMM body: block = 128 threads (4 waves), block tile
// M=128 x N=64, K stepped by 32.  Each wave owns 32 rows (2 A fragments,
// 8 f32 accumulators) so every LDS-staged B fragment feeds 2 WMMAs.
template <int KDIM>
static __device__ __forceinline__ void gemm_tile(const _Float16* __restrict__ A,
                                                 const _Float16* __restrict__ Bt,
                                                 _Float16* sbuf,
                                                 int m0, int n0, v8f acc[2][4]) {
  const int lane = threadIdx.x & 31;
  const int m = lane & 15, g = lane >> 4;
  const _Float16* arow0 = A + (size_t)(m0 + m) * KDIM;
  const _Float16* arow1 = A + (size_t)(m0 + 16 + m) * KDIM;

  for (int kc = 0; kc < KDIM; kc += 32) {
    stage_chunk(sbuf, Bt + (size_t)n0 * KDIM, KDIM, kc);
    __syncthreads();
    v16h a0 = load_a_frag(arow0, kc, g);
    v16h a1 = load_a_frag(arow1, kc, g);
    if (kc + 32 < KDIM) {
      __builtin_prefetch(arow0 + kc + 32, 0, 1);
      __builtin_prefetch(arow1 + kc + 32, 0, 1);
    }
#pragma unroll
    for (int t = 0; t < 4; ++t) {
      v16h bf = load_b_frag_lds(sbuf + (16 * t + m) * LPAD, g);
      acc[0][t] = wmma_f16(a0, bf, acc[0][t]);
      acc[1][t] = wmma_f16(a1, bf, acc[1][t]);
    }
    __syncthreads();
  }
}

// Kernel 4: MLP layer 1: H = GELU(head_out @ W1 + b1), f16 out.
__global__ void mlp1_kernel(const _Float16* __restrict__ A,
                            const _Float16* __restrict__ W1t,
                            const float* __restrict__ b1,
                            _Float16* __restrict__ Hout) {
  __shared__ __align__(32) _Float16 sbuf[64 * LPAD];
  const int wave = threadIdx.x >> 5, lane = threadIdx.x & 31;
  const int m = lane & 15, g = lane >> 4;
  const int m0 = blockIdx.x * 128 + wave * 32;
  const int n0 = blockIdx.y * 64;

  v8f acc[2][4];
#pragma unroll
  for (int f = 0; f < 2; ++f)
#pragma unroll
    for (int t = 0; t < 4; ++t)
#pragma unroll
      for (int r = 0; r < 8; ++r) acc[f][t][r] = 0.0f;

  gemm_tile<EE>(A, W1t, sbuf, m0, n0, acc);

#pragma unroll
  for (int t = 0; t < 4; ++t) {
    const int n = n0 + 16 * t + m;
    const float bias = b1[n];
#pragma unroll
    for (int f = 0; f < 2; ++f)
#pragma unroll
      for (int r = 0; r < 8; ++r) {
        const int row = m0 + 16 * f + r + 8 * g;
        const float v = acc[f][t][r] + bias;
        const float gel = 0.5f * v * (1.0f + erff(v * 0.70710678118654752440f));
        Hout[(size_t)row * FFF + n] = (_Float16)gel;
      }
  }
}

// Kernel 5: MLP layer 2 + residual: out = H @ W2 + b2 + input1, f32 out.
__global__ void mlp2_kernel(const _Float16* __restrict__ Hh,
                            const _Float16* __restrict__ W2t,
                            const float* __restrict__ b2,
                            const float* __restrict__ x,
                            float* __restrict__ out) {
  __shared__ __align__(32) _Float16 sbuf[64 * LPAD];
  const int wave = threadIdx.x >> 5, lane = threadIdx.x & 31;
  const int m = lane & 15, g = lane >> 4;
  const int m0 = blockIdx.x * 128 + wave * 32;
  const int n0 = blockIdx.y * 64;

  v8f acc[2][4];
#pragma unroll
  for (int f = 0; f < 2; ++f)
#pragma unroll
    for (int t = 0; t < 4; ++t)
#pragma unroll
      for (int r = 0; r < 8; ++r) acc[f][t][r] = 0.0f;

  gemm_tile<FFF>(Hh, W2t, sbuf, m0, n0, acc);

#pragma unroll
  for (int t = 0; t < 4; ++t) {
    const int n = n0 + 16 * t + m;
    const float bias = b2[n];
#pragma unroll
    for (int f = 0; f < 2; ++f)
#pragma unroll
      for (int r = 0; r < 8; ++r) {
        const int row = m0 + 16 * f + r + 8 * g;
        out[(size_t)row * EE + n] = acc[f][t][r] + bias + x[(size_t)row * EE + n];
      }
  }
}

// ---------------------------------------------------------------------------
extern "C" void kernel_launch(void* const* d_in, const int* in_sizes, int n_in,
                              void* d_out, int out_size, void* d_ws, size_t ws_size,
                              hipStream_t stream) {
  const float* input1 = (const float*)d_in[0];
  const float* Wv = (const float*)d_in[1];
  const float* bv = (const float*)d_in[2];
  const float* Wq = (const float*)d_in[3];
  const float* bq = (const float*)d_in[4];
  const float* Wk = (const float*)d_in[5];
  const float* bk = (const float*)d_in[6];
  const float* W1 = (const float*)d_in[7];
  const float* b1 = (const float*)d_in[8];
  const float* W2 = (const float*)d_in[9];
  const float* b2 = (const float*)d_in[10];
  float* out = (float*)d_out;

  char* ws = (char*)d_ws;
  // Workspace layout (all 256B aligned):
  float*    Qf  = (float*)(ws + 0);                        // B*H*S f32      = 256 KB
  float*    Kf  = (float*)(ws + (256u << 10));             // B*H*S f32      = 256 KB
  _Float16* Vt  = (_Float16*)(ws + (512u << 10));          // B*H*D*S f16    = 8 MB
  _Float16* HO  = (_Float16*)(ws + (512u << 10) + 8388608u);   // head_out f16 = 8 MB
  _Float16* W1t = (_Float16*)(ws + (512u << 10) + 16777216u);  // 2 MB
  _Float16* W2t = (_Float16*)(ws + (512u << 10) + 18874368u);  // 2 MB
  _Float16* Hm  = (_Float16*)(ws + (512u << 10) + 20971520u);  // NTOK*FF f16 = 32 MB

  // 1) weight convert/transpose
  convert_weights<<<(EE * FFF + 255) / 256, 256, 0, stream>>>(W1, W2, W1t, W2t);
  // 2) layernorm + projections
  ln_qkv_kernel<<<NTOK, 256, 0, stream>>>(input1, Wv, bv, Wq, bq, Wk, bk, Qf, Kf, Vt);
  // 3) flash attention (WMMA P@V, LDS-staged V chunks)
  attn_kernel<<<dim3(SS / 64, BB * HH), 128, 0, stream>>>(Qf, Kf, Vt, HO);
  // 4) MLP layer 1 (WMMA GEMM + GELU)
  mlp1_kernel<<<dim3(NTOK / 128, FFF / 64), 128, 0, stream>>>(HO, W1t, b1, Hm);
  // 5) MLP layer 2 (WMMA GEMM + bias + residual)
  mlp2_kernel<<<dim3(NTOK / 128, EE / 64), 128, 0, stream>>>(Hm, W2t, b2, input1, out);
}
